// NGP_42082089566816
// MI455X (gfx1250) — matrix operand order repaired and checked by reading, not compiled, not run on hardware
//
#include <hip/hip_runtime.h>

typedef __attribute__((ext_vector_type(16))) _Float16 v16h;
typedef __attribute__((ext_vector_type(8)))  _Float16 v8h;
typedef __attribute__((ext_vector_type(8)))  float    v8f;

#define TSIZE 524288
#define TMASK (TSIZE - 1)
#define ROWLD 72                 // padded row stride (halfs) for activation tiles
#define BUFSZ (32 * ROWLD)

// LDS weight offsets (halfs). Weights stored PRE-SWIZZLED into WMMA B-fragment
// layout: sequence of 32x16 K-tiles; each tile = 512 halfs as two 256-half
// blocks; within a block lane l owns 8 contiguous halfs (16B).
//   element(lane l, block b, j) = W[kb*32 + b*16 + (l>>4)*8 + j][nb*16 + (l&15)]
#define W1OFF 0                  // K32  x N64  -> 4 tiles
#define W2OFF 2048               // K64  x N16  -> 2 tiles
#define W3OFF 3072               // K64* x N64  -> 8 tiles (*rows 43..63 zeroed)
#define W4OFF 7168               // K64  x N64  -> 8 tiles
#define W5OFF 11264              // K64  x N16* -> 2 tiles (*cols 3..15 zeroed)
#define WTOT  12288
// bias offsets (floats)
#define B1OFF 0
#define B2OFF 64
#define B3OFF 80
#define B4OFF 144
#define B5OFF 208
#define BTOT  224

static __device__ __forceinline__ v8f wmma16(v16h a, v16h b, v8f c) {
  // D = A(16x32 f16) x B(32x16 f16) + C(16x16 f32)
  return __builtin_amdgcn_wmma_f32_16x16x32_f16(false, a, false, b, (short)0, c,
                                                false, false);
}

// A-fragment: 16x32 f16 tile from row-major LDS [16][ld], K window [k0, k0+32)
// ISA layout: lane m (0..15): K {0..7, 16..23}; lane m+16: K {8..15, 24..31}
static __device__ __forceinline__ v16h frag_a(const _Float16* base, int ld, int k0) {
  int lane = threadIdx.x & 31;
  int m = lane & 15;
  int hi = lane >> 4;
  const _Float16* row = base + m * ld + k0 + hi * 8;
  v8h lo = *(const v8h*)(row);        // K k0 + hi*8 + 0..7
  v8h hv = *(const v8h*)(row + 16);   // K k0 + 16 + hi*8 + 0..7
  v16h a;
#pragma unroll
  for (int j = 0; j < 8; ++j) { a[j] = lo[j]; a[8 + j] = hv[j]; }
  return a;
}

// B-fragment from pre-swizzled LDS weights: two conflict-free 16B loads.
static __device__ __forceinline__ v16h frag_b(const _Float16* base, int tile) {
  int lane = threadIdx.x & 31;
  const _Float16* t = base + tile * 512 + lane * 8;
  v8h lo = *(const v8h*)(t);          // slots j=0..7
  v8h hv = *(const v8h*)(t + 256);    // slots j=8..15
  v16h b;
#pragma unroll
  for (int j = 0; j < 8; ++j) { b[j] = lo[j]; b[8 + j] = hv[j]; }
  return b;
}

// Stage a weight matrix W[Kreal][Nreal] (row-major f32) into swizzled f16 tiles,
// zero-padding to Kpad x Npad.  ntnShift = log2(Npad/16).
static __device__ __forceinline__ void stage_swz(_Float16* dst, const float* W,
                                                 int Kreal, int Nreal,
                                                 int Kpad, int Npad,
                                                 int ntnShift, int tid) {
  int total = Kpad * Npad;
  for (int i = tid; i < total; i += 64) {
    int tile = i >> 9;
    int rem  = i & 511;
    int blk  = rem >> 8;
    int l    = (rem >> 3) & 31;
    int jj   = rem & 7;
    int kb = tile >> ntnShift;
    int nb = tile - (kb << ntnShift);
    int K = kb * 32 + blk * 16 + (l >> 4) * 8 + jj;
    int c = nb * 16 + (l & 15);
    float v = (K < Kreal && c < Nreal) ? W[K * Nreal + c] : 0.0f;
    dst[i] = (_Float16)v;
  }
}

// C/D layout: VGPR r -> row (r + 8*hi), col = lane&15
static __device__ __forceinline__ void store_relu(v8f c, const float* bias, int n0,
                                                  _Float16* dst, int ld) {
  int lane = threadIdx.x & 31;
  int n = lane & 15;
  int hi = lane >> 4;
#pragma unroll
  for (int r = 0; r < 8; ++r) {
    float v = c[r] + bias[n0 + n];
    v = fmaxf(v, 0.0f);
    dst[(r + hi * 8) * ld + n0 + n] = (_Float16)v;
  }
}

__global__ __launch_bounds__(64) void ngp_fused(
    const float* __restrict__ x, const float* __restrict__ dirs,
    const float* __restrict__ tables,
    const float* __restrict__ w1, const float* __restrict__ b1,
    const float* __restrict__ w2, const float* __restrict__ b2,
    const float* __restrict__ w3, const float* __restrict__ b3,
    const float* __restrict__ w4, const float* __restrict__ b4,
    const float* __restrict__ w5, const float* __restrict__ b5,
    float* __restrict__ out, int N)
{
  __shared__ __align__(16) _Float16 sW[WTOT];
  __shared__ float sB[BTOT];
  __shared__ __align__(16) _Float16 sAct[2 * 3 * BUFSZ]; // per wave: buf0, buf1, xi
  __shared__ unsigned char sMask[64];

  const int tid  = threadIdx.x;
  const int lane = tid & 31;
  const int wv   = tid >> 5;
  const int row  = tid & 31;

  // ---------- stage weights to LDS as f16, pre-swizzled to B-fragment layout ----------
  stage_swz(sW + W1OFF, w1, 32, 64, 32, 64, 2, tid);
  stage_swz(sW + W2OFF, w2, 64, 16, 64, 16, 0, tid);
  stage_swz(sW + W3OFF, w3, 43, 64, 64, 64, 2, tid);
  stage_swz(sW + W4OFF, w4, 64, 64, 64, 64, 2, tid);
  stage_swz(sW + W5OFF, w5, 64,  3, 64, 16, 0, tid);
  sB[B1OFF + tid] = b1[tid];
  sB[B3OFF + tid] = b3[tid];
  sB[B4OFF + tid] = b4[tid];
  if (tid < 16) sB[B2OFF + tid] = b2[tid];
  if (tid < 16) sB[B5OFF + tid] = (tid < 3) ? b5[tid] : 0.0f;

  _Float16* buf0 = sAct + wv * 3 * BUFSZ;   // feats -> later c1 (color L1 out)
  _Float16* buf1 = buf0 + BUFSZ;            // h1    -> later c2
  _Float16* xib  = buf0 + 2 * BUFSZ;        // xi[64] per row

  // ---------- per-lane hash-grid encode (gathers stay in 192MB L2) ----------
  const int p = blockIdx.x * 64 + tid;
  const float inv3 = 1.0f / 3.0f;   // 1/AABB_SCALE
  float xs0 = x[p * 3 + 0] * inv3;
  float xs1 = x[p * 3 + 1] * inv3;
  float xs2 = x[p * 3 + 2] * inv3;
  bool inside = (fabsf(xs0) < 0.5f) && (fabsf(xs1) < 0.5f) && (fabsf(xs2) < 0.5f);
  sMask[tid] = inside ? 1 : 0;
  const float hi1 = 1.0f - 1e-7f;
  float p0 = fminf(fmaxf(xs0 + 0.5f, 0.0f), hi1);
  float p1 = fminf(fmaxf(xs1 + 0.5f, 0.0f), hi1);
  float p2 = fminf(fmaxf(xs2 + 0.5f, 0.0f), hi1);

  _Float16* feat = buf0 + row * ROWLD;
  const float NLf[16] = {16.f, 22.f, 30.f, 42.f, 58.f, 80.f, 110.f, 152.f,
                         210.f, 290.f, 400.f, 552.f, 762.f, 1052.f, 1452.f, 2004.f};
#pragma unroll
  for (int i = 0; i < 16; ++i) {
    float s0 = p0 * NLf[i], s1 = p1 * NLf[i], s2 = p2 * NLf[i];
    float f0 = floorf(s0), f1 = floorf(s1), f2 = floorf(s2);
    float t0 = s0 - f0, t1 = s1 - f1, t2 = s2 - f2;
    unsigned bx = (unsigned)f0, by = (unsigned)f1, bz = (unsigned)f2;
    const float2* tab = (const float2*)(tables + (size_t)i * (TSIZE * 2));
    float a0 = 0.0f, a1 = 0.0f;
#pragma unroll
    for (int c = 0; c < 8; ++c) {
      unsigned vx = bx + (c & 1);
      unsigned vy = by + ((c >> 1) & 1);
      unsigned vz = bz + ((c >> 2) & 1);
      unsigned h = vx ^ (vy * 2654435761u) ^ (vz * 805459861u);
      float2 g = tab[h & TMASK];
      float wx = (c & 1) ? t0 : 1.0f - t0;
      float wy = ((c >> 1) & 1) ? t1 : 1.0f - t1;
      float wz = ((c >> 2) & 1) ? t2 : 1.0f - t2;
      float w = wx * wy * wz;
      a0 += w * g.x;
      a1 += w * g.y;
    }
    feat[2 * i]     = (_Float16)a0;
    feat[2 * i + 1] = (_Float16)a1;
  }

  // ---------- direction encoding into xi cols 0..26; zero pad 43..63 ----------
  {
    _Float16* xr = xib + row * ROWLD;
    float d0 = dirs[p * 3 + 0], d1 = dirs[p * 3 + 1], d2 = dirs[p * 3 + 2];
    xr[0] = (_Float16)d0; xr[1] = (_Float16)d1; xr[2] = (_Float16)d2;
#pragma unroll
    for (int k = 0; k < 4; ++k) {
      float fr = (float)(1 << k);
      xr[3 + 0 * 4 + k]  = (_Float16)__sinf(d0 * fr);
      xr[3 + 1 * 4 + k]  = (_Float16)__sinf(d1 * fr);
      xr[3 + 2 * 4 + k]  = (_Float16)__sinf(d2 * fr);
      xr[15 + 0 * 4 + k] = (_Float16)__cosf(d0 * fr);
      xr[15 + 1 * 4 + k] = (_Float16)__cosf(d1 * fr);
      xr[15 + 2 * 4 + k] = (_Float16)__cosf(d2 * fr);
    }
#pragma unroll
    for (int c = 43; c < 64; ++c) xr[c] = (_Float16)0.0f;
  }
  __syncthreads();

  // ---------- L1: feats[16x32] @ w1[32x64] -> buf1 (relu) ----------
  {
    v16h aA = frag_a(buf0, ROWLD, 0);
    v16h aB = frag_a(buf0 + 16 * ROWLD, ROWLD, 0);
#pragma unroll
    for (int nb = 0; nb < 4; ++nb) {
      v16h b = frag_b(sW + W1OFF, nb);            // kb=0
      v8f c0 = {}; c0 = wmma16(aA, b, c0);
      store_relu(c0, sB + B1OFF, nb * 16, buf1, ROWLD);
      v8f c1 = {}; c1 = wmma16(aB, b, c1);
      store_relu(c1, sB + B1OFF, nb * 16, buf1 + 16 * ROWLD, ROWLD);
    }
  }
  __syncthreads();

  // ---------- L2: h1[16x64] @ w2[64x16] -> h; h -> xi cols 27..42, log_sigma out ----------
  {
    v16h b0 = frag_b(sW + W2OFF, 0);              // kb=0
    v16h b1f = frag_b(sW + W2OFF, 1);             // kb=1
    const int n = lane & 15, hb = lane >> 4;
#pragma unroll
    for (int t = 0; t < 2; ++t) {
      v16h a0 = frag_a(buf1 + t * 16 * ROWLD, ROWLD, 0);
      v16h a1 = frag_a(buf1 + t * 16 * ROWLD, ROWLD, 32);
      v8f c = {};
      c = wmma16(a0, b0, c);
      c = wmma16(a1, b1f, c);
#pragma unroll
      for (int r = 0; r < 8; ++r) {
        int rr = r + hb * 8;
        float v = c[r] + sB[B2OFF + n];
        xib[(t * 16 + rr) * ROWLD + 27 + n] = (_Float16)v;
        if (n == 0) {
          int rowi = wv * 32 + t * 16 + rr;
          int pp = blockIdx.x * 64 + rowi;
          out[(size_t)3 * N + pp] = sMask[rowi] ? v : -100000.0f;
        }
      }
    }
  }
  __syncthreads();

  // ---------- L3: xi[16x64] @ w3[64x64] -> buf0 (relu) ----------
#pragma unroll
  for (int t = 0; t < 2; ++t) {
    v16h a0 = frag_a(xib + t * 16 * ROWLD, ROWLD, 0);
    v16h a1 = frag_a(xib + t * 16 * ROWLD, ROWLD, 32);
#pragma unroll
    for (int nb = 0; nb < 4; ++nb) {
      v8f c = {};
      c = wmma16(a0, frag_b(sW + W3OFF, 0 * 4 + nb), c);
      c = wmma16(a1, frag_b(sW + W3OFF, 1 * 4 + nb), c);
      store_relu(c, sB + B3OFF, nb * 16, buf0 + t * 16 * ROWLD, ROWLD);
    }
  }
  __syncthreads();

  // ---------- L4: c1[16x64] @ w4[64x64] -> buf1 (relu) ----------
#pragma unroll
  for (int t = 0; t < 2; ++t) {
    v16h a0 = frag_a(buf0 + t * 16 * ROWLD, ROWLD, 0);
    v16h a1 = frag_a(buf0 + t * 16 * ROWLD, ROWLD, 32);
#pragma unroll
    for (int nb = 0; nb < 4; ++nb) {
      v8f c = {};
      c = wmma16(a0, frag_b(sW + W4OFF, 0 * 4 + nb), c);
      c = wmma16(a1, frag_b(sW + W4OFF, 1 * 4 + nb), c);
      store_relu(c, sB + B4OFF, nb * 16, buf1 + t * 16 * ROWLD, ROWLD);
    }
  }
  __syncthreads();

  // ---------- L5: c2[16x64] @ w5[64x16 padded] -> sigmoid color ----------
  {
    v16h b0 = frag_b(sW + W5OFF, 0);
    v16h b1f = frag_b(sW + W5OFF, 1);
    const int n = lane & 15, hb = lane >> 4;
#pragma unroll
    for (int t = 0; t < 2; ++t) {
      v16h a0 = frag_a(buf1 + t * 16 * ROWLD, ROWLD, 0);
      v16h a1 = frag_a(buf1 + t * 16 * ROWLD, ROWLD, 32);
      v8f c = {};
      c = wmma16(a0, b0, c);
      c = wmma16(a1, b1f, c);
      if (n < 3) {
#pragma unroll
        for (int r = 0; r < 8; ++r) {
          int rr = r + hb * 8;
          int rowi = wv * 32 + t * 16 + rr;
          int pp = blockIdx.x * 64 + rowi;
          float v = c[r] + sB[B5OFF + n];
          v = 1.0f / (1.0f + __expf(-v));
          out[(size_t)pp * 3 + n] = sMask[rowi] ? v : 0.0f;
        }
      }
    }
  }
}

extern "C" void kernel_launch(void* const* d_in, const int* in_sizes, int n_in,
                              void* d_out, int out_size, void* d_ws, size_t ws_size,
                              hipStream_t stream) {
  const float* x   = (const float*)d_in[0];
  const float* dir = (const float*)d_in[1];
  const float* tab = (const float*)d_in[2];
  const float* w1  = (const float*)d_in[3];
  const float* b1  = (const float*)d_in[4];
  const float* w2  = (const float*)d_in[5];
  const float* b2  = (const float*)d_in[6];
  const float* w3  = (const float*)d_in[7];
  const float* b3  = (const float*)d_in[8];
  const float* w4  = (const float*)d_in[9];
  const float* b4  = (const float*)d_in[10];
  const float* w5  = (const float*)d_in[11];
  const float* b5  = (const float*)d_in[12];
  float* out = (float*)d_out;

  int N = in_sizes[0] / 3;         // 1048576
  int blocks = N / 64;             // 64 points per block (2 waves x 32)
  hipLaunchKernelGGL(ngp_fused, dim3(blocks), dim3(64), 0, stream,
                     x, dir, tab, w1, b1, w2, b2, w3, b3, w4, b4, w5, b5, out, N);
}
